// MultiConvAttn_7533372637190
// MI455X (gfx1250) — matrix operand order repaired and checked by reading, not compiled, not run on hardware
//
#include <hip/hip_runtime.h>
#include <hip/hip_bf16.h>

// ---------------- types ----------------
typedef unsigned short u16;
typedef __attribute__((ext_vector_type(16))) __bf16 v16bf;
typedef __attribute__((ext_vector_type(8)))  float  v8f;
typedef __attribute__((ext_vector_type(4)))  unsigned int u4;
typedef __attribute__((ext_vector_type(4)))  float  f4;

struct Q2 { u4 a, b; };

__device__ __forceinline__ u16 f2bf(float f) {
    unsigned int u = __float_as_uint(f);
    unsigned int r = (u + 0x7fffu + ((u >> 16) & 1u)) >> 16;
    return (u16)r;
}
__device__ __forceinline__ float bf2f(u16 s) {
    return __uint_as_float(((unsigned int)s) << 16);
}

// A fragment (16x32 bf16, M=16 pixels, K=32): lane<16 -> K 0..7 & 16..23,
// lane>=16 -> K 8..15 & 24..31 (base already offset by half*8 elements).
__device__ __forceinline__ v16bf ldfragA(const u16* base) {
    Q2 q;
    q.a = *(const u4*)(base);        // K = half*8 + 0..7
    q.b = *(const u4*)(base + 16);   // K = 16 + half*8 + 0..7
    return __builtin_bit_cast(v16bf, q);
}
// B fragment: pre-packed per-lane 16 contiguous bf16 (32 bytes).
__device__ __forceinline__ v16bf ldfragB(const u16* base) {
    Q2 q;
    q.a = *(const u4*)(base);
    q.b = *(const u4*)(base + 8);
    return __builtin_bit_cast(v16bf, q);
}

// ---------------- problem constants ----------------
// seq (B=4,T=16,C=128,48,48); BT=64; padded NHWC 50x50x128 bf16.
#define NB   4
#define NT   16
#define NBT  64
#define NC   128
#define NH   48
#define NW   48
#define NHP  50
#define NWP  50
#define NPIX 2304          // 48*48
#define HEADS 4
#define DH   32

// ---------------- kernel 0: zero ONLY pad borders of xpad+kpad ------------
// 196 border pixels per 50x50 frame, 64 frames, 128 ch, 2 buffers = 6.4 MB
// (vs 82 MB full memset). Interiors are fully rewritten every call.
__global__ void k_zero_border(u16* __restrict__ xpad, u16* __restrict__ kpad) {
    int tid = blockIdx.x * blockDim.x + threadIdx.x;
    if (tid >= NBT * 196) return;
    int bt = tid / 196, i = tid % 196;
    int yy, xx;
    if (i < 50)       { yy = 0;          xx = i;       }
    else if (i < 100) { yy = 49;         xx = i - 50;  }
    else if (i < 148) { yy = i - 100 + 1; xx = 0;      }
    else              { yy = i - 148 + 1; xx = 49;     }
    size_t base = (((size_t)bt * NHP + yy) * NWP + xx) * NC;
    u4 z = {0u, 0u, 0u, 0u};
    u4* px = (u4*)(xpad + base);
    u4* pk = (u4*)(kpad + base);
#pragma unroll
    for (int j = 0; j < 16; ++j) { px[j] = z; pk[j] = z; }
}

// ---------------- kernel 1: NCHW f32 -> padded NHWC bf16 (LDS transpose) ---
__global__ void __launch_bounds__(256) k_pack_x(const float* __restrict__ x,
                                                u16* __restrict__ xpad) {
    __shared__ float tile[32][33];
    int bt = blockIdx.z, ct = blockIdx.y, pt = blockIdx.x;  // 64 x 4 x 72
    int p0 = pt * 32, c0 = ct * 32;
#pragma unroll
    for (int i = 0; i < 4; ++i) {
        int c = c0 + threadIdx.y + i * 8;
        int p = p0 + threadIdx.x;
        tile[threadIdx.y + i * 8][threadIdx.x] =
            x[((size_t)bt * NC + c) * NPIX + p];
    }
    __syncthreads();
#pragma unroll
    for (int i = 0; i < 4; ++i) {
        int p = p0 + threadIdx.y + i * 8;
        int c = c0 + threadIdx.x;
        int yy = p / NW + 1, xx = p % NW + 1;
        xpad[(((size_t)bt * NHP + yy) * NWP + xx) * NC + c] =
            f2bf(tile[threadIdx.x][threadIdx.y + i * 8]);
    }
}

// ---------------- kernel 2: pack W2 + proj_w into B-fragment order --------
// B 32x16 bf16 layout: lane = n + 16*((k>>3)&1), element = (k&7) + 8*(k>>4).
__global__ void k_pack_w(const float* __restrict__ w2,
                         const float* __restrict__ pw,
                         u16* __restrict__ bp, u16* __restrict__ pp) {
    int tid = blockIdx.x * blockDim.x + threadIdx.x;
    if (tid < 128 * 1152) {                       // W2: (co,ci,3,3)
        int co = tid / 1152, rem = tid % 1152;
        int tap = rem / 128, ci = rem % 128;
        float w = w2[((size_t)co * 128 + ci) * 9 + tap];
        int ntile = co >> 4, n = co & 15, kc = ci >> 5, k = ci & 31;
        int frag = (ntile * 9 + tap) * 4 + kc;
        int lane = n + (((k >> 3) & 1) << 4);
        bp[((size_t)frag * 32 + lane) * 16 + ((k & 7) | ((k >> 4) << 3))] = f2bf(w);
    } else if (tid < 128 * 1152 + 128 * 128) {    // proj: (co,ci,1,1)
        int t = tid - 128 * 1152;
        int co = t >> 7, ci = t & 127;
        float w = pw[(size_t)(co << 7) + ci];
        int ntile = co >> 4, n = co & 15, kc = ci >> 5, k = ci & 31;
        int frag = ntile * 4 + kc;
        int lane = n + (((k >> 3) & 1) << 4);
        pp[((size_t)frag * 32 + lane) * 16 + ((k & 7) | ((k >> 4) << 3))] = f2bf(w);
    }
}

// ---------------- kernel 3: K = conv3x3(x,W2)+b2 via WMMA bf16 ------------
// block = (frame bt, output row y), 8 waves = 8 N-tiles of 16 channels,
// 3 M-subtiles of 16 pixels, K-loop = 9 taps x 4 chunks of 32 ch.
__global__ void __launch_bounds__(256) k_conv(const u16* __restrict__ xpad,
                                              const u16* __restrict__ bp,
                                              const float* __restrict__ bias,
                                              u16* __restrict__ kpad) {
    int bt = blockIdx.x / NH;
    int y  = blockIdx.x % NH;
    int wv   = threadIdx.x >> 5;           // N-tile
    int lane = threadIdx.x & 31;
    int m = lane & 15, half = lane >> 4;
    int co = wv * 16 + m;                  // N index of this lane's C/D column
    float bv = bias[co];
    const u16* bpw = bp + (size_t)(wv * 36) * 512;   // 36 frags * 512 elems
    for (int msub = 0; msub < 3; ++msub) {
        int x0 = msub * 16;
        v8f acc = {};
        for (int tap = 0; tap < 9; ++tap) {
            int dy = tap / 3, dx = tap % 3;          // padded offsets 0..2
            const u16* arow = xpad +
                (((size_t)bt * NHP + (y + dy)) * NWP + (x0 + m + dx)) * NC +
                half * 8;
            const u16* brow = bpw + (size_t)(tap * 4) * 512 + lane * 16;
#pragma unroll
            for (int kc = 0; kc < 4; ++kc) {
                v16bf a = ldfragA(arow + kc * 32);
                v16bf b = ldfragB(brow + (size_t)kc * 512);
                acc = __builtin_amdgcn_wmma_f32_16x16x32_bf16(
                    false, a, false, b, (short)0, acc, false, false);
            }
        }
#pragma unroll
        for (int r = 0; r < 8; ++r) {
            float v = acc[r] + bv;
            int xo = x0 + r + 8 * half;    // D: M = r + 8*half
            kpad[(((size_t)bt * NHP + (y + 1)) * NWP + (xo + 1)) * NC + co] =
                f2bf(v);
        }
    }
}

// ---------------- kernel 4: SK -> softmax_ti -> Obar ----------------------
// A = softmax_ti(SK) is independent of tk (SQ cancels), so O is shared by
// all 16 output frames. One thread per (b, pixel, head).
__global__ void __launch_bounds__(256) k_attn(const u16* __restrict__ kpad,
                                              const float* __restrict__ w3k,
                                              const float* __restrict__ w3b,
                                              u16* __restrict__ obar) {
    int tid = blockIdx.x * blockDim.x + threadIdx.x;
    if (tid >= NB * NPIX * HEADS) return;
    int h = tid & 3;
    int p = (tid >> 2) % NPIX;
    int b = tid / (NPIX * HEADS);
    int y = p / NW, x = p % NW;
    const size_t fstride = (size_t)NHP * NWP * NC;   // frame stride in kpad
    float s[NT];
    float bias = w3b[h];
#pragma unroll
    for (int ti = 0; ti < NT; ++ti) s[ti] = bias;
    for (int tap = 0; tap < 9; ++tap) {
        int dy = tap / 3, dx = tap % 3;
        const u16* kb0 = kpad + ((size_t)(b * NT) * NHP + (y + dy)) * NWP * NC +
                         (size_t)(x + dx) * NC + h * DH;
        for (int c = 0; c < DH; ++c) {
            float w = w3k[(size_t)(h * DH + c) * 9 + tap];
            const u16* kb = kb0 + c;
#pragma unroll
            for (int ti = 0; ti < NT; ++ti)
                s[ti] += w * bf2f(kb[(size_t)ti * fstride]);
        }
    }
    float mx = s[0];
#pragma unroll
    for (int i = 1; i < NT; ++i) mx = fmaxf(mx, s[i]);
    float sum = 0.f;
#pragma unroll
    for (int i = 0; i < NT; ++i) { s[i] = __expf(s[i] - mx); sum += s[i]; }
    float inv = 1.f / sum;
    const u16* kc0 = kpad + ((size_t)(b * NT) * NHP + (y + 1)) * NWP * NC +
                     (size_t)(x + 1) * NC + h * DH;
    u16* ob = obar + ((size_t)b * NPIX + p) * NC + h * DH;
    for (int d = 0; d < DH; ++d) {
        float o = 0.f;
#pragma unroll
        for (int ti = 0; ti < NT; ++ti)
            o += s[ti] * bf2f(kc0[(size_t)ti * fstride + d]);
        ob[d] = f2bf(o * inv);
    }
}

// ---------------- kernel 5: out = proj(Obar)+pb, broadcast over 16 t ------
__global__ void __launch_bounds__(256) k_proj(const u16* __restrict__ obar,
                                              const u16* __restrict__ pp,
                                              const float* __restrict__ pb,
                                              float* __restrict__ out) {
    __shared__ float stg[8][16][17];
    int b = blockIdx.x / NH;
    int y = blockIdx.x % NH;
    int wv = threadIdx.x >> 5;
    int lane = threadIdx.x & 31;
    int m = lane & 15, half = lane >> 4;
    float bv = pb[wv * 16 + m];
    const u16* ppw = pp + (size_t)(wv * 4) * 512;
    for (int msub = 0; msub < 3; ++msub) {
        int x0 = msub * 16;
        v8f acc = {};
        const u16* arow = obar +
            ((size_t)b * NPIX + y * NW + x0 + m) * NC + half * 8;
#pragma unroll
        for (int kc = 0; kc < 4; ++kc) {
            v16bf a  = ldfragA(arow + kc * 32);
            v16bf bf = ldfragB(ppw + (size_t)kc * 512 + lane * 16);
            acc = __builtin_amdgcn_wmma_f32_16x16x32_bf16(
                false, a, false, bf, (short)0, acc, false, false);
        }
        __syncthreads();
#pragma unroll
        for (int r = 0; r < 8; ++r)
            stg[wv][m][r + 8 * half] = acc[r] + bv;   // [co_local][x_local]
        __syncthreads();
        int col = lane >> 1;        // co_local 0..15
        int xp  = lane & 1;         // x half 0/1
        f4 v0 = { stg[wv][col][xp * 8 + 0], stg[wv][col][xp * 8 + 1],
                  stg[wv][col][xp * 8 + 2], stg[wv][col][xp * 8 + 3] };
        f4 v1 = { stg[wv][col][xp * 8 + 4], stg[wv][col][xp * 8 + 5],
                  stg[wv][col][xp * 8 + 6], stg[wv][col][xp * 8 + 7] };
        int co = wv * 16 + col;
        for (int t = 0; t < NT; ++t) {
            float* dst = out +
                ((((size_t)(b * NT + t)) * NC + co) * NH + y) * NW +
                x0 + xp * 8;
            *(f4*)dst = v0;
            *(f4*)(dst + 4) = v1;
        }
    }
}

// ---------------- launch ----------------
extern "C" void kernel_launch(void* const* d_in, const int* in_sizes, int n_in,
                              void* d_out, int out_size, void* d_ws, size_t ws_size,
                              hipStream_t stream) {
    const float* seq    = (const float*)d_in[0];
    // d_in[1] (W1_w), d_in[2] (W1_b), d_in[5] (W3q_w) are mathematically dead:
    // softmax over ti of SQ[tk]+SK[ti] == softmax(SK[ti]).
    const float* W2_w   = (const float*)d_in[3];
    const float* W2_b   = (const float*)d_in[4];
    const float* W3k_w  = (const float*)d_in[6];
    const float* W3_b   = (const float*)d_in[7];
    const float* proj_w = (const float*)d_in[8];
    const float* proj_b = (const float*)d_in[9];
    float* out = (float*)d_out;

    char* ws = (char*)d_ws;
    const size_t SZ_PAD = (size_t)NBT * NHP * NWP * NC * 2;   // 40,960,000 B
    u16* xpad = (u16*)(ws);
    u16* kpad = (u16*)(ws + SZ_PAD);
    u16* bp   = (u16*)(ws + 2 * SZ_PAD);                      // 294,912 B
    u16* pp   = (u16*)(ws + 2 * SZ_PAD + 294912);             //  32,768 B
    u16* obar = (u16*)(ws + 2 * SZ_PAD + 294912 + 32768);     // 2,359,296 B

    {   // zero only the pad borders (interiors fully rewritten below)
        int n = NBT * 196;
        k_zero_border<<<(n + 255) / 256, 256, 0, stream>>>(xpad, kpad);
    }
    k_pack_x<<<dim3(72, 4, 64), dim3(32, 8), 0, stream>>>(seq, xpad);
    {
        int n = 128 * 1152 + 128 * 128;
        k_pack_w<<<(n + 255) / 256, 256, 0, stream>>>(W2_w, proj_w, bp, pp);
    }
    k_conv<<<NBT * NH, 256, 0, stream>>>(xpad, bp, W2_b, kpad);
    {
        int n = NB * NPIX * HEADS;
        k_attn<<<(n + 255) / 256, 256, 0, stream>>>(kpad, W3k_w, W3_b, obar);
    }
    k_proj<<<NB * NH, 256, 0, stream>>>(obar, pp, proj_b, out);
}
